// DistanceAttention_15607911154066
// MI455X (gfx1250) — compile-verified, matrix operand order checked by reading, not run on hardware
//
#include <hip/hip_runtime.h>
#include <hip/hip_bf16.h>

typedef _Float16 v16h __attribute__((ext_vector_type(16)));
typedef float    v8f  __attribute__((ext_vector_type(8)));

#define B_  8
#define S_  1024
#define D_  512
#define H_  8
#define DH  64

// ---------------------------------------------------------------------------
// Fragment-layout index helpers (ISA 7.12.2, 16-bit operands, wave32).
// A-fragment: lane = hi*16 + m,  slot i <-> k = (i&7) + ((i>>3)<<4) + 8*hi
// B-fragment: lane = hi*16 + n,  slot i <-> k (same map)
// Inverse:    hi = (k>>3)&1,     slot = (k&7) + ((k>>4)<<3)
// ---------------------------------------------------------------------------
__device__ __forceinline__ void k_inv(int kc, int& hi, int& slot) {
    hi   = (kc >> 3) & 1;
    slot = (kc & 7) + ((kc >> 4) << 3);
}

// A-layout over an [M x 512] activation matrix: [rowtile][chunk16][lane32][slot16]
__device__ __forceinline__ size_t a_frag_idx(int row, int k) {
    int mt = row >> 4, m = row & 15;
    int chunk = k >> 5, hi, slot;
    k_inv(k & 31, hi, slot);
    return (((size_t)(mt * 16 + chunk) * 32 + ((hi << 4) + m)) << 4) + slot;
}

// B-layout over a [512 x 512] weight matrix: [chunk16][ntile32][lane32][slot16]
__device__ __forceinline__ size_t b_frag_idx(int k, int n) {
    int chunk = k >> 5, hi, slot;
    k_inv(k & 31, hi, slot);
    return (((size_t)(chunk * 32 + (n >> 4)) * 32 + ((hi << 4) + (n & 15))) << 4) + slot;
}

// Per-(b,h) layout used for BOTH Q (A-frags) and K (B-frags of K^T):
// [stile64][chunk2][lane32][slot16]  (65536 halfs per head)
__device__ __forceinline__ size_t qk_frag_idx(int bh, int s, int d) {
    int hi, slot;
    k_inv(d & 31, hi, slot);
    return (size_t)bh * 65536 +
           ((((size_t)(s >> 4) * 2 + (d >> 5)) * 32 + ((hi << 4) + (s & 15))) << 4) + slot;
}

// Per-(b,h) V layout: B-frags for context GEMM: [kchunk32][ntile4][lane32][slot16]
__device__ __forceinline__ size_t v_frag_idx(int bh, int s, int d) {
    int hi, slot;
    k_inv(s & 31, hi, slot);
    return (size_t)bh * 65536 +
           ((((size_t)(s >> 5) * 4 + (d >> 4)) * 32 + ((hi << 4) + (d & 15))) << 4) + slot;
}

__device__ __forceinline__ v8f wmma_f16(v16h a, v16h b, v8f c) {
    return __builtin_amdgcn_wmma_f32_16x16x32_f16(
        false, a, false, b, (short)0, c, false, false);
}

union pk2 { _Float16 h[2]; unsigned int u; };

// ---------------------------------------------------------------------------
// Pack activations f32 -> f16 A-fragment layout. One k-pair per thread:
// coalesced float2 read, single packed 4B write (slots are k-contiguous).
// ---------------------------------------------------------------------------
__global__ __launch_bounds__(256)
void pack_a_kernel(const float* __restrict__ x0, const float* __restrict__ x1,
                   const float* __restrict__ x2,
                   _Float16* __restrict__ f0, _Float16* __restrict__ f1,
                   _Float16* __restrict__ f2) {
    const float* X; _Float16* F;
    if (blockIdx.z == 0)      { X = x0; F = f0; }
    else if (blockIdx.z == 1) { X = x1; F = f1; }
    else                      { X = x2; F = f2; }
    int pi  = blockIdx.x * 256 + threadIdx.x;     // 0 .. 2M-1 pairs
    int row = pi >> 8;                            // 256 pairs per 512-wide row
    int k   = (pi & 255) << 1;
    float2 v = *(const float2*)&X[(size_t)row * D_ + k];
    pk2 p; p.h[0] = (_Float16)v.x; p.h[1] = (_Float16)v.y;
    *(unsigned int*)&F[a_frag_idx(row, k)] = p.u;
}

// Pack weights f32 -> f16 B-fragment layout (one element/thread, coalesced read)
__global__ __launch_bounds__(256)
void pack_w_kernel(const float* __restrict__ w0, const float* __restrict__ w1,
                   const float* __restrict__ w2, const float* __restrict__ w3,
                   _Float16* __restrict__ f0, _Float16* __restrict__ f1,
                   _Float16* __restrict__ f2, _Float16* __restrict__ f3) {
    const float* W; _Float16* F;
    if (blockIdx.z == 0)      { W = w0; F = f0; }
    else if (blockIdx.z == 1) { W = w1; F = f1; }
    else if (blockIdx.z == 2) { W = w2; F = f2; }
    else                      { W = w3; F = f3; }
    int ei = blockIdx.x * 256 + threadIdx.x;      // 0 .. 262143
    int k = ei >> 9, n = ei & 511;
    F[b_frag_idx(k, n)] = (_Float16)W[(size_t)ei];
}

// ---------------------------------------------------------------------------
// Q/K/V projection GEMM, LDS-free: wave = 16x64 tile, fragments are
// contiguous per-lane v16h loads (b128 pairs). Output scattered into the
// attention-ready fragment layouts.
// ---------------------------------------------------------------------------
__global__ __launch_bounds__(256)
void proj_kernel(const _Float16* __restrict__ xfq, const _Float16* __restrict__ xfk,
                 const _Float16* __restrict__ xfv,
                 const _Float16* __restrict__ wfq, const _Float16* __restrict__ wfk,
                 const _Float16* __restrict__ wfv,
                 _Float16* __restrict__ qf, _Float16* __restrict__ kf,
                 _Float16* __restrict__ vf) {
    const _Float16* XF; const _Float16* WF; _Float16* O;
    if (blockIdx.z == 0)      { XF = xfq; WF = wfq; O = qf; }
    else if (blockIdx.z == 1) { XF = xfk; WF = wfk; O = kf; }
    else                      { XF = xfv; WF = wfv; O = vf; }

    const int tid = threadIdx.x, lane = tid & 31, w = tid >> 5;
    const int hi = lane >> 4, ln = lane & 15;
    const int mt   = blockIdx.x * 8 + w;          // 0..511 (16-row tiles)
    const int col0 = blockIdx.y * 64;
    const int nt0  = col0 >> 4;

    v8f acc[4] = {};
#pragma unroll
    for (int chunk = 0; chunk < 16; ++chunk) {
        v16h a = *(const v16h*)(XF + (((size_t)(mt * 16 + chunk) * 32 + lane) << 4));
#pragma unroll
        for (int ns = 0; ns < 4; ++ns) {
            v16h b = *(const v16h*)(WF + (((size_t)(chunk * 32 + nt0 + ns) * 32 + lane) << 4));
            acc[ns] = wmma_f16(a, b, acc[ns]);
        }
    }

#pragma unroll
    for (int ns = 0; ns < 4; ++ns) {
#pragma unroll
        for (int r = 0; r < 8; ++r) {
            int grow = mt * 16 + r + hi * 8;
            int gcol = col0 + ns * 16 + ln;
            int bidx = grow >> 10, s = grow & 1023;
            int h    = gcol >> 6,  d = gcol & 63;
            int bh   = bidx * H_ + h;
            _Float16 val = (_Float16)acc[ns][r];
            if (blockIdx.z == 2) O[v_frag_idx(bh, s, d)]  = val;
            else                 O[qk_frag_idx(bh, s, d)] = val;
        }
    }
}

// ---------------------------------------------------------------------------
// Fused attention per (b, h, 16-row q tile): WMMA scores -> dist*scale ->
// mask -> f32 softmax in LDS -> attn stream-out -> WMMA context.
// ---------------------------------------------------------------------------
__global__ __launch_bounds__(128)
void attn_kernel(const _Float16* __restrict__ qf, const _Float16* __restrict__ kf,
                 const _Float16* __restrict__ vf,
                 const float* __restrict__ dist, const unsigned char* __restrict__ mask,
                 float* __restrict__ attn_out, _Float16* __restrict__ ctxf) {
    __shared__ float    sc[16][S_];          // 64 KB exact softmax strip
    __shared__ _Float16 pf[32 * 32 * 16];    // 32 KB probs, A-frag layout [kc][lane][slot]

    const int tid = threadIdx.x, lane = tid & 31, w = tid >> 5;
    const int hi = lane >> 4, ln = lane & 15;
    const int qt = blockIdx.x, h = blockIdx.y, b = blockIdx.z;
    const int q0 = qt * 16;
    const int bh = b * H_ + h;

    const size_t hb    = (size_t)bh * 65536;
    const size_t dbase = ((size_t)b * S_ + q0) * S_;
    const float  scale = 0.125f;

    // Q fragments: contiguous per-lane 32B loads
    v16h a0 = *(const v16h*)(qf + hb + (((size_t)(qt * 2 + 0) * 32 + lane) << 4));
    v16h a1 = *(const v16h*)(qf + hb + (((size_t)(qt * 2 + 1) * 32 + lane) << 4));

    // --- scores: wave w owns every 4th 16-column tile ---
    for (int kt = w; kt < 64; kt += 4) {
        v16h b0 = *(const v16h*)(kf + hb + (((size_t)(kt * 2 + 0) * 32 + lane) << 4));
        v16h b1 = *(const v16h*)(kf + hb + (((size_t)(kt * 2 + 1) * 32 + lane) << 4));
        v8f c = {};
        c = wmma_f16(a0, b0, c);
        c = wmma_f16(a1, b1, c);
        const int k0 = kt * 16;
#pragma unroll
        for (int r = 0; r < 8; ++r) {
            int m = r + hi * 8, kcol = k0 + ln;
            float v = c[r] * scale;
            v *= dist[dbase + (size_t)m * S_ + kcol];
            if (mask[dbase + (size_t)m * S_ + kcol]) v = -1e10f;
            sc[m][kcol] = v;
        }
    }
    __syncthreads();

    // --- softmax: 8 lanes per row, butterfly reduce in lane-octet ---
    const int row = tid >> 3, sub = tid & 7;
    float mx = -3.0e38f;
    for (int k = sub; k < S_; k += 8) mx = fmaxf(mx, sc[row][k]);
#pragma unroll
    for (int off = 4; off; off >>= 1) mx = fmaxf(mx, __shfl_xor(mx, off, 8));

    float sum = 0.f;
    for (int k = sub; k < S_; k += 8) {
        float e = __expf(sc[row][k] - mx);
        sc[row][k] = e;
        sum += e;
    }
#pragma unroll
    for (int off = 4; off; off >>= 1) sum += __shfl_xor(sum, off, 8);
    const float inv = 1.0f / sum;

    // normalize: stream attn to HBM (float2) + packed f16 A-frag copy to LDS
    float* arow = attn_out + (((size_t)bh * S_) + (q0 + row)) * S_;
    for (int k = sub * 2; k < S_; k += 16) {
        float v0 = sc[row][k] * inv;
        float v1 = sc[row][k + 1] * inv;
        *(float2*)&arow[k] = make_float2(v0, v1);
        int kc = k >> 5, hi2, slot;
        k_inv(k & 31, hi2, slot);
        pk2 p; p.h[0] = (_Float16)v0; p.h[1] = (_Float16)v1;
        *(unsigned int*)&pf[(((kc * 32) + ((hi2 << 4) + row)) << 4) + slot] = p.u;
    }
    __syncthreads();

    // --- context: C[16x64] = P[16x1024] @ V[1024x64]; wave w = n-tile w ---
    v8f acc = {};
#pragma unroll
    for (int kc = 0; kc < 32; ++kc) {
        v16h af = *(const v16h*)&pf[((size_t)(kc * 32 + lane)) << 4];
        v16h bf = *(const v16h*)(vf + hb + (((size_t)(kc * 4 + w) * 32 + lane) << 4));
        acc = wmma_f16(af, bf, acc);
    }
#pragma unroll
    for (int r = 0; r < 8; ++r) {
        int q   = q0 + r + hi * 8;
        int col = h * 64 + w * 16 + ln;
        ctxf[a_frag_idx(b * S_ + q, col)] = (_Float16)acc[r];
    }
}

// ---------------------------------------------------------------------------
// output = ctx @ W_O + input_Q ; LDS-free fragment GEMM, f32 out.
// ---------------------------------------------------------------------------
__global__ __launch_bounds__(256)
void out_kernel(const _Float16* __restrict__ ctxf, const _Float16* __restrict__ wfo,
                const float* __restrict__ resid, float* __restrict__ out) {
    const int tid = threadIdx.x, lane = tid & 31, w = tid >> 5;
    const int hi = lane >> 4, ln = lane & 15;
    const int mt   = blockIdx.x * 8 + w;
    const int col0 = blockIdx.y * 64;
    const int nt0  = col0 >> 4;

    v8f acc[4] = {};
#pragma unroll
    for (int chunk = 0; chunk < 16; ++chunk) {
        v16h a = *(const v16h*)(ctxf + (((size_t)(mt * 16 + chunk) * 32 + lane) << 4));
#pragma unroll
        for (int ns = 0; ns < 4; ++ns) {
            v16h b = *(const v16h*)(wfo + (((size_t)(chunk * 32 + nt0 + ns) * 32 + lane) << 4));
            acc[ns] = wmma_f16(a, b, acc[ns]);
        }
    }
#pragma unroll
    for (int ns = 0; ns < 4; ++ns) {
#pragma unroll
        for (int r = 0; r < 8; ++r) {
            int grow = mt * 16 + r + hi * 8;
            int gcol = col0 + ns * 16 + ln;
            out[(size_t)grow * D_ + gcol] = acc[ns][r] + resid[(size_t)grow * D_ + gcol];
        }
    }
}

// ---------------------------------------------------------------------------
extern "C" void kernel_launch(void* const* d_in, const int* in_sizes, int n_in,
                              void* d_out, int out_size, void* d_ws, size_t ws_size,
                              hipStream_t stream) {
    (void)in_sizes; (void)n_in; (void)out_size; (void)ws_size;

    const float*         input_Q = (const float*)d_in[0];
    const float*         input_K = (const float*)d_in[1];
    const float*         input_V = (const float*)d_in[2];
    const float*         dist    = (const float*)d_in[3];
    const unsigned char* mask    = (const unsigned char*)d_in[4];
    const float*         W_Q     = (const float*)d_in[5];
    const float*         W_K     = (const float*)d_in[6];
    const float*         W_V     = (const float*)d_in[7];
    const float*         W_O     = (const float*)d_in[8];

    float* out  = (float*)d_out;                 // [B,S,D]
    float* attn = out + (size_t)B_ * S_ * D_;    // [B,H,S,S]

    _Float16* ws = (_Float16*)d_ws;
    const size_t actElems  = (size_t)B_ * S_ * D_;   // 4,194,304 halfs
    const size_t wElems    = (size_t)D_ * D_;        //   262,144 halfs
    _Float16* xfq = ws;
    _Float16* xfk = xfq + actElems;
    _Float16* xfv = xfk + actElems;
    _Float16* wfq = xfv + actElems;
    _Float16* wfk = wfq + wElems;
    _Float16* wfv = wfk + wElems;
    _Float16* wfo = wfv + wElems;
    _Float16* qf  = wfo + wElems;
    _Float16* kf  = qf  + actElems;
    _Float16* vf  = kf  + actElems;
    _Float16* ctx = vf  + actElems;

    pack_a_kernel<<<dim3(8192, 1, 3), 256, 0, stream>>>(
        input_Q, input_K, input_V, xfq, xfk, xfv);
    pack_w_kernel<<<dim3(1024, 1, 4), 256, 0, stream>>>(
        W_Q, W_K, W_V, W_O, wfq, wfk, wfv, wfo);
    proj_kernel<<<dim3(64, 8, 3), 256, 0, stream>>>(
        xfq, xfk, xfv, wfq, wfk, wfv, qf, kf, vf);
    attn_kernel<<<dim3(64, H_, B_), 128, 0, stream>>>(
        qf, kf, vf, dist, mask, attn, ctx);
    out_kernel<<<dim3(64, 8), 256, 0, stream>>>(ctx, wfo, input_Q, out);
}
